// GAT_Transformer_88897233093106
// MI455X (gfx1250) — matrix operand order repaired and checked by reading, not compiled, not run on hardware
//
#include <hip/hip_runtime.h>
#include <hip/hip_bf16.h>

// MI455X (gfx1250): GAT (edge-softmax message passing) + 2-layer transformer.
// Dense GEMMs via v_wmma_f32_16x16x32_bf16, wave32, 16x64 tile per wave.
// All GEMM operands are bf16 in memory (A mirrors maintained by producers),
// so the inner loop is pure vector loads + WMMA (no conversion, no guards).

typedef __bf16 bf16;
typedef __attribute__((ext_vector_type(16))) __bf16 v16bf;
typedef __attribute__((ext_vector_type(8)))  __bf16 v8bf;
typedef __attribute__((ext_vector_type(8)))  float  v8f;

#define DEV static __device__ __forceinline__

DEV bf16 f2bf(float f) {
  unsigned u = __builtin_bit_cast(unsigned, f);
  unsigned r = u + 0x7FFFu + ((u >> 16) & 1u);   // round-to-nearest-even
  return __builtin_bit_cast(bf16, (unsigned short)(r >> 16));
}
DEV bf16 bf0() { return __builtin_bit_cast(bf16, (unsigned short)0); }
DEV float bf2f(bf16 b) {
  unsigned u = (unsigned)__builtin_bit_cast(unsigned short, b) << 16;
  return __builtin_bit_cast(float, u);
}
DEV float lrelu(float x) { return x > 0.f ? x : 0.2f * x; }
// monotonic uint encoding of float for atomicMax-based segment max
DEV unsigned enc_f(float f) {
  unsigned u = __builtin_bit_cast(unsigned, f);
  return (u & 0x80000000u) ? ~u : (u | 0x80000000u);
}
DEV float dec_f(unsigned e) {
  unsigned u = (e & 0x80000000u) ? (e & 0x7FFFFFFFu) : ~e;
  return __builtin_bit_cast(float, u);
}
#define ENC_NEG_INF 0x007FFFFFu   // enc(-inf)

// ---------------------------------------------------------------- utilities
__global__ void k_zero(float* p, long n) {
  long i = (long)blockIdx.x * 256 + threadIdx.x;
  if (i < n) p[i] = 0.f;
}
__global__ void k_fillu(unsigned* p, unsigned v, long n) {
  long i = (long)blockIdx.x * 256 + threadIdx.x;
  if (i < n) p[i] = v;
}
// W [K,N] f32 row-major -> out [N,Kpad] bf16 (transposed, K zero-padded)
__global__ void k_convW(const float* __restrict__ W, bf16* __restrict__ out,
                        int K, int N, int Kpad) {
  int i = blockIdx.x * 256 + threadIdx.x;
  if (i >= N * Kpad) return;
  int n = i / Kpad, k = i - n * Kpad;
  out[i] = (k < K) ? f2bf(W[(size_t)k * N + n]) : bf0();
}
// A [M,K] f32 -> O [M,Kp] bf16 (row zero-padded to Kp)
__global__ void k_cvt_pad(const float* __restrict__ A, bf16* __restrict__ O,
                          int K, int Kp, long MKp) {
  long i = (long)blockIdx.x * 256 + threadIdx.x;
  if (i >= MKp) return;
  int k = (int)(i % Kp);
  long m = i / Kp;
  O[i] = (k < K) ? f2bf(A[m * K + k]) : bf0();
}
// fold attention vector: u[f*4+h] = sum_c W[f*256 + h*64 + c] * att[h*64+c]
__global__ void k_pack_att(const float* __restrict__ W, const float* __restrict__ att,
                           float* __restrict__ u, int F) {
  int i = blockIdx.x * 256 + threadIdx.x;
  if (i >= F * 4) return;
  int f = i >> 2, h = i & 3;
  const float* wr = W + (size_t)f * 256 + h * 64;
  const float* ar = att + h * 64;
  float s = 0.f;
  for (int c = 0; c < 64; ++c) s += wr[c] * ar[c];
  u[i] = s;
}

// ---------------------------------------------------------------- WMMA GEMM
// D[M,N] = A[M,lda](bf16) * Bw[N,Kpad](bf16, transposed); K % 32 == 0.
// epi: 0=none 1=+bias 2=+bias,relu. Writes f32 C and/or bf16 Cb (non-null ones).
__global__ void __launch_bounds__(256) k_gemm(
    const bf16* __restrict__ A, int lda, const bf16* __restrict__ Bw,
    float* __restrict__ C, bf16* __restrict__ Cb, const float* __restrict__ bias,
    int M, int N, int K, int Kpad, int nT, int epi) {
  int wid  = blockIdx.x * 8 + (threadIdx.x >> 5);
  int lane = threadIdx.x & 31;
  int mT = (M + 15) >> 4;
  if (wid >= mT * nT) return;
  int m0 = (wid / nT) << 4;
  int n0 = (wid % nT) << 6;

  v8f c0 = {}, c1 = {}, c2 = {}, c3 = {};
  int mrow = m0 + (lane & 15);
  if (mrow >= M) mrow = M - 1;              // branchless clamp (stores are guarded)
  int kAo = (lane & 16) ? 16 : 0;           // A: lanes 16-31 cover K 16-31
  int kBo = (lane & 16) ? 16 : 0;           // B: lanes 16-31 cover K 16-31
  const bf16* ap  = A + (size_t)mrow * lda + kAo;
  const bf16* bp0 = Bw + (size_t)(n0 + (lane & 15)) * Kpad + kBo;

#pragma unroll 2
  for (int kb = 0; kb < K; kb += 32) {
    v8bf alo = *(const v8bf*)(ap + kb);
    v8bf ahi = *(const v8bf*)(ap + kb + 8);
    v16bf af = __builtin_shufflevector(alo, ahi, 0, 1, 2, 3, 4, 5, 6, 7,
                                       8, 9, 10, 11, 12, 13, 14, 15);
    __builtin_prefetch((const void*)(ap + kb + 32), 0, 0);
    const bf16* bp = bp0 + kb;
    v16bf b0 = *(const v16bf*)(bp);
    v16bf b1 = *(const v16bf*)(bp + (size_t)16 * Kpad);
    v16bf b2 = *(const v16bf*)(bp + (size_t)32 * Kpad);
    v16bf b3 = *(const v16bf*)(bp + (size_t)48 * Kpad);
    c0 = __builtin_amdgcn_wmma_f32_16x16x32_bf16(false, af, false, b0, (short)0, c0, false, false);
    c1 = __builtin_amdgcn_wmma_f32_16x16x32_bf16(false, af, false, b1, (short)0, c1, false, false);
    c2 = __builtin_amdgcn_wmma_f32_16x16x32_bf16(false, af, false, b2, (short)0, c2, false, false);
    c3 = __builtin_amdgcn_wmma_f32_16x16x32_bf16(false, af, false, b3, (short)0, c3, false, false);
  }
  int nc = n0 + (lane & 15);
#pragma unroll
  for (int j = 0; j < 8; ++j) {
    int m = m0 + ((lane & 16) ? j + 8 : j);
    if (m >= M) continue;
    size_t o = (size_t)m * N + nc;
    float v0 = c0[j], v1 = c1[j], v2 = c2[j], v3 = c3[j];
    if (epi >= 1) { v0 += bias[nc]; v1 += bias[nc + 16]; v2 += bias[nc + 32]; v3 += bias[nc + 48]; }
    if (epi == 2) { v0 = fmaxf(v0, 0.f); v1 = fmaxf(v1, 0.f); v2 = fmaxf(v2, 0.f); v3 = fmaxf(v3, 0.f); }
    if (C)  { C[o] = v0; C[o + 16] = v1; C[o + 32] = v2; C[o + 48] = v3; }
    if (Cb) { Cb[o] = f2bf(v0); Cb[o + 16] = f2bf(v1); Cb[o + 32] = f2bf(v2); Cb[o + 48] = f2bf(v3); }
  }
}

// ---------------------------------------------------------------- GAT kernels
__global__ void k_degsum(const int* __restrict__ dst, const float* __restrict__ te,
                         float* __restrict__ sum, float* __restrict__ deg, int E) {
  long i = (long)blockIdx.x * 256 + threadIdx.x;
  if (i >= (long)E * 64) return;
  int e = (int)(i >> 6), c = (int)(i & 63);
  int d = dst[e];
  atomicAdd(&sum[(size_t)d * 64 + c], te[i]);
  if (c == 0) atomicAdd(&deg[d], 1.f);
}
__global__ void k_mean(float* __restrict__ sum, const float* __restrict__ deg, int Nn) {
  int i = blockIdx.x * 256 + threadIdx.x;
  if (i >= Nn * 64) return;
  sum[i] /= fmaxf(deg[i >> 6], 1.f);
}
__global__ void k_node_att(const float* __restrict__ x, const float* __restrict__ us,
                           const float* __restrict__ ud, float* __restrict__ a_src,
                           float* __restrict__ a_dst, int Nn) {
  int i = blockIdx.x * 256 + threadIdx.x;
  if (i >= Nn * 4) return;
  int n = i >> 2, h = i & 3;
  const float* xr = x + (size_t)n * 256;
  float s = 0.f, d = 0.f;
  for (int f = 0; f < 256; ++f) { float xv = xr[f]; s += xv * us[f * 4 + h]; d += xv * ud[f * 4 + h]; }
  a_src[i] = s; a_dst[i] = d;
}
__global__ void k_a_edge(const float* __restrict__ te, const float* __restrict__ meanE,
                         const float* __restrict__ we, float* __restrict__ a_edge,
                         int E, int E2) {
  int e = blockIdx.x * 256 + threadIdx.x;
  if (e >= E2) return;
  const float* r = (e < E) ? te + (size_t)e * 64 : meanE + (size_t)(e - E) * 64;
  float s0 = 0, s1 = 0, s2 = 0, s3 = 0;
  for (int c = 0; c < 64; ++c) {
    float t = r[c]; const float* w = we + c * 4;
    s0 += t * w[0]; s1 += t * w[1]; s2 += t * w[2]; s3 += t * w[3];
  }
  float* o = a_edge + (size_t)e * 4;
  o[0] = s0; o[1] = s1; o[2] = s2; o[3] = s3;
}
__global__ void k_alpha_max(const int* __restrict__ ei, const float* __restrict__ a_src,
                            const float* __restrict__ a_dst, const float* __restrict__ a_edge,
                            unsigned* __restrict__ amax, int E, int E2) {
  long i = (long)blockIdx.x * 256 + threadIdx.x;
  if (i >= (long)E2 * 4) return;
  int e = (int)(i >> 2), h = (int)(i & 3);
  int s = (e < E) ? ei[e] : (e - E);
  int d = (e < E) ? ei[E + e] : (e - E);
  float al = lrelu(a_src[s * 4 + h] + a_dst[d * 4 + h] + a_edge[i]);
  atomicMax(&amax[d * 4 + h], enc_f(al));
}
__global__ void k_denom(const int* __restrict__ ei, const float* __restrict__ a_src,
                        const float* __restrict__ a_dst, const float* __restrict__ a_edge,
                        const unsigned* __restrict__ amax, float* __restrict__ denom,
                        int E, int E2) {
  long i = (long)blockIdx.x * 256 + threadIdx.x;
  if (i >= (long)E2 * 4) return;
  int e = (int)(i >> 2), h = (int)(i & 3);
  int s = (e < E) ? ei[e] : (e - E);
  int d = (e < E) ? ei[E + e] : (e - E);
  unsigned enc = amax[d * 4 + h];
  float am = (enc == ENC_NEG_INF) ? 0.f : dec_f(enc);
  float al = lrelu(a_src[s * 4 + h] + a_dst[d * 4 + h] + a_edge[i]);
  atomicAdd(&denom[d * 4 + h], expf(al - am));
}
__global__ void k_scatter(const int* __restrict__ ei, const float* __restrict__ a_src,
                          const float* __restrict__ a_dst, const float* __restrict__ a_edge,
                          const unsigned* __restrict__ amax, const float* __restrict__ denom,
                          const float* __restrict__ xp, float* __restrict__ acc,
                          int E, int E2) {
  long i = (long)blockIdx.x * 256 + threadIdx.x;
  if (i >= (long)E2 * 64) return;
  int e = (int)(i >> 6), c = (int)(i & 63);
  int s = (e < E) ? ei[e] : (e - E);
  int d = (e < E) ? ei[E + e] : (e - E);
#pragma unroll
  for (int h = 0; h < 4; ++h) {
    unsigned enc = amax[d * 4 + h];
    float am = (enc == ENC_NEG_INF) ? 0.f : dec_f(enc);
    float al = lrelu(a_src[s * 4 + h] + a_dst[d * 4 + h] + a_edge[(size_t)e * 4 + h]);
    float w  = expf(al - am) / (denom[d * 4 + h] + 1e-16f);
    atomicAdd(&acc[(size_t)d * 256 + h * 64 + c], xp[(size_t)s * 256 + h * 64 + c] * w);
  }
}
__global__ void k_finalize(const float* __restrict__ acc, const float* __restrict__ bias,
                           float* __restrict__ g, int Nn) {
  int i = blockIdx.x * 256 + threadIdx.x;
  if (i >= Nn * 64) return;
  int n = i >> 6, c = i & 63;
  const float* a = acc + (size_t)n * 256 + c;
  g[i] = fmaxf(0.25f * (a[0] + a[64] + a[128] + a[192]) + bias[c], 0.f);
}

// ---------------------------------------------------------------- transformer
__global__ void k_comb(const float* __restrict__ seq, const int* __restrict__ nidx,
                       const float* __restrict__ g1, const float* __restrict__ g2,
                       bf16* __restrict__ comb, int Bn) {
  long i = (long)blockIdx.x * 256 + threadIdx.x;
  if (i >= (long)Bn * 48 * 160) return;
  int f = (int)(i % 160);
  long ts = i / 160;
  int b = (int)(ts / 48), s = (int)(ts % 48);
  float v;
  if (f < 32) v = seq[((size_t)b * 48 + s) * 32 + f];
  else {
    int ni = nidx[b];
    v = (f < 96) ? g1[(size_t)ni * 64 + (f - 32)] : g2[(size_t)ni * 64 + (f - 96)];
  }
  comb[i] = f2bf(v);
}
__global__ void k_addpos(float* __restrict__ h, bf16* __restrict__ hb,
                         const float* __restrict__ pos, int T) {
  long i = (long)blockIdx.x * 256 + threadIdx.x;
  if (i >= (long)T * 256) return;
  int tok = (int)(i >> 8), d = (int)(i & 255);
  float v = h[i] + pos[(tok % 48) * 256 + d];
  h[i] = v;
  hb[i] = f2bf(v);
}
// per (seq-in-chunk, head) attention; S=48, dh=32; bf16 in, bf16 out
__global__ void __launch_bounds__(64) k_attn(const bf16* __restrict__ qm,
                                             const bf16* __restrict__ km,
                                             const bf16* __restrict__ vm,
                                             bf16* __restrict__ ctx) {
  int bc = blockIdx.x >> 3, h = blockIdx.x & 7;
  __shared__ float Ks[48][32];
  __shared__ float Vs[48][32];
  int tid = threadIdx.x;
  for (int i = tid; i < 48 * 32; i += 64) {
    int s = i >> 5, c = i & 31;
    size_t o = ((size_t)bc * 48 + s) * 256 + h * 32 + c;
    Ks[s][c] = bf2f(km[o]);
    Vs[s][c] = bf2f(vm[o]);
  }
  __syncthreads();
  if (tid < 48) {
    size_t qo = ((size_t)bc * 48 + tid) * 256 + h * 32;
    float qr[32];
#pragma unroll
    for (int c = 0; c < 32; ++c) qr[c] = bf2f(qm[qo + c]);
    float sc[48];
    float mx = -3.0e38f;
    for (int j = 0; j < 48; ++j) {
      float s = 0.f;
#pragma unroll
      for (int c = 0; c < 32; ++c) s += qr[c] * Ks[j][c];
      s *= 0.1767766953f;  // 1/sqrt(32)
      sc[j] = s;
      mx = fmaxf(mx, s);
    }
    float sum = 0.f;
    for (int j = 0; j < 48; ++j) { float e = expf(sc[j] - mx); sc[j] = e; sum += e; }
    float inv = 1.f / sum;
    float o[32];
#pragma unroll
    for (int c = 0; c < 32; ++c) o[c] = 0.f;
    for (int j = 0; j < 48; ++j) {
      float w = sc[j] * inv;
#pragma unroll
      for (int c = 0; c < 32; ++c) o[c] += w * Vs[j][c];
    }
#pragma unroll
    for (int c = 0; c < 32; ++c) ctx[qo + c] = f2bf(o[c]);
  }
}
// h = LN(h + add) * s + b ; one wave per 256-wide row; also writes bf16 mirror
__global__ void k_add_ln(float* __restrict__ h, bf16* __restrict__ hb,
                         const float* __restrict__ add,
                         const float* __restrict__ sc, const float* __restrict__ bi, int T) {
  int row  = blockIdx.x * 8 + (threadIdx.x >> 5);
  int lane = threadIdx.x & 31;
  if (row >= T) return;
  float* hp = h + (size_t)row * 256;
  bf16*  bp = hb + (size_t)row * 256;
  const float* ap = add + (size_t)row * 256;
  float x[8];
  float sum = 0.f;
#pragma unroll
  for (int i = 0; i < 8; ++i) { x[i] = hp[lane + 32 * i] + ap[lane + 32 * i]; sum += x[i]; }
  for (int o = 16; o; o >>= 1) sum += __shfl_xor(sum, o, 32);
  float mu = sum * (1.f / 256.f);
  float var = 0.f;
#pragma unroll
  for (int i = 0; i < 8; ++i) { float d = x[i] - mu; var += d * d; }
  for (int o = 16; o; o >>= 1) var += __shfl_xor(var, o, 32);
  float rinv = rsqrtf(var * (1.f / 256.f) + 1e-5f);
#pragma unroll
  for (int i = 0; i < 8; ++i) {
    int c = lane + 32 * i;
    float v = (x[i] - mu) * rinv * sc[c] + bi[c];
    hp[c] = v;
    bp[c] = f2bf(v);
  }
}
__global__ void k_out(const float* __restrict__ h, const float* __restrict__ oW,
                      const float* __restrict__ ob, float* __restrict__ out, int nb, int b0) {
  int r = blockIdx.x * 8 + (threadIdx.x >> 5);
  int lane = threadIdx.x & 31;
  if (r >= nb) return;
  const float* hp = h + ((size_t)r * 48 + 47) * 256;
  float s = 0.f;
  for (int i = lane; i < 256; i += 32) s += hp[i] * oW[i];
  for (int o = 16; o; o >>= 1) s += __shfl_xor(s, o, 32);
  if (lane == 0) out[b0 + r] = s + ob[0];
}

// ---------------------------------------------------------------- host
static inline int g1d(long n) { return (int)((n + 255) / 256); }

static void gemm(hipStream_t st, const bf16* A, int lda, const bf16* Bw,
                 float* C, bf16* Cb, const float* bias,
                 int M, int N, int K, int Kpad, int epi) {
  long waves = (long)((M + 15) / 16) * (N / 64);
  k_gemm<<<(int)((waves + 7) / 8), 256, 0, st>>>(A, lda, Bw, C, Cb, bias,
                                                 M, N, K, Kpad, N / 64, epi);
}

extern "C" void kernel_launch(void* const* d_in, const int* in_sizes, int n_in,
                              void* d_out, int out_size, void* d_ws, size_t ws_size,
                              hipStream_t stream) {
  (void)n_in; (void)out_size;
  const float* seq  = (const float*)d_in[0];
  const int*   ei   = (const int*)d_in[1];   // [2,E]: src = ei, dst = ei+E
  const float* ea   = (const float*)d_in[2];
  const float* xnf  = (const float*)d_in[3];
  const int*   nidx = (const int*)d_in[4];
  const int E  = in_sizes[1] / 2;
  const int Nn = in_sizes[3] / 256;
  const int Bn = in_sizes[4];
  const int E2 = E + Nn;

  // params leaves in jax pytree (sorted-key) order
#define P(i) ((const float*)d_in[i])
  const float* edge_tr_W = P(5);
  const float* edge_tr_b = P(6);
  const float* att_dst[2]  = { P(7),  P(13) };
  const float* att_edge[2] = { P(8),  P(14) };
  const float* att_src[2]  = { P(9),  P(15) };
  const float* gbias[2]    = { P(10), P(16) };
  const float* lin_W[2]    = { P(11), P(17) };
  const float* lin_eW[2]   = { P(12), P(18) };
  const float* in_W = P(19);
  const float* in_b = P(20);
  struct Lay {
    const float *Wk, *Wo, *Wq, *Wv, *bk, *bo, *bq, *bv;
    const float *f1W, *f1b, *f2W, *f2b, *ln1b, *ln1s, *ln2b, *ln2s;
  } L[2];
  for (int l = 0; l < 2; ++l) {
    int b = 21 + 16 * l;
    L[l] = { P(b+0), P(b+1), P(b+2), P(b+3), P(b+4), P(b+5), P(b+6), P(b+7),
             P(b+8), P(b+9), P(b+10), P(b+11), P(b+12), P(b+13), P(b+14), P(b+15) };
  }
  const float* out_W = P(53);
  const float* out_b = P(54);
  const float* pos   = P(55);
#undef P

  // ------------------------------------------------------------ workspace
  size_t off = 0;
  char* base = (char*)d_ws;
  auto carve = [&](size_t bytes) -> char* {
    char* p = base + off;
    off += (bytes + 255) & ~(size_t)255;
    return p;
  };
  bf16* wb_te     = (bf16*)carve(64 * 32 * 2);
  bf16* wb_lin[2] = { (bf16*)carve(256 * 256 * 2), (bf16*)carve(256 * 256 * 2) };
  bf16* wb_in     = (bf16*)carve(256 * 160 * 2);
  bf16 *wb_q[2], *wb_k[2], *wb_v[2], *wb_o[2], *wb_f1[2], *wb_f2[2];
  for (int l = 0; l < 2; ++l) {
    wb_q[l]  = (bf16*)carve(256 * 256 * 2);
    wb_k[l]  = (bf16*)carve(256 * 256 * 2);
    wb_v[l]  = (bf16*)carve(256 * 256 * 2);
    wb_o[l]  = (bf16*)carve(256 * 256 * 2);
    wb_f1[l] = (bf16*)carve(1024 * 256 * 2);
    wb_f2[l] = (bf16*)carve(256 * 1024 * 2);
  }
  float *usrc[2], *udst[2], *wed[2];
  for (int g = 0; g < 2; ++g) {
    usrc[g] = (float*)carve(256 * 4 * 4);
    udst[g] = (float*)carve(256 * 4 * 4);
    wed[g]  = (float*)carve(64 * 4 * 4);
  }
  float* gout[2] = { (float*)carve((size_t)Nn * 64 * 4), (float*)carve((size_t)Nn * 64 * 4) };
  bf16* combbf = (bf16*)carve((size_t)Bn * 48 * 160 * 2);

  const int TMAX = 256 * 48;
  size_t gat_bytes = 4 * ((size_t)E * 64 + (size_t)Nn * 64 + Nn + (size_t)E2 * 4
                          + (size_t)Nn * 16 + (size_t)Nn * 512)
                   + 2 * ((size_t)E * 32 + (size_t)Nn * 256);
  size_t tr_bytes  = (size_t)TMAX * (256 * 4 * 2 + (256 * 5 + 1024) * 2);
  size_t arena_b = gat_bytes > tr_bytes ? gat_bytes : tr_bytes;
  char* ar = carve(arena_b);
  if (off > ws_size) return;  // workspace too small; bail (no unsafe writes)

  // GAT-phase view of arena
  float*    te     = (float*)ar;
  float*    meanE  = te + (size_t)E * 64;
  float*    deg    = meanE + (size_t)Nn * 64;
  float*    a_edge = deg + Nn;
  float*    a_src  = a_edge + (size_t)E2 * 4;
  float*    a_dst  = a_src + (size_t)Nn * 4;
  unsigned* amax   = (unsigned*)(a_dst + (size_t)Nn * 4);
  float*    denom  = (float*)(amax + (size_t)Nn * 4);
  float*    xp     = denom + (size_t)Nn * 4;
  float*    accb   = xp + (size_t)Nn * 256;
  bf16*     eabf   = (bf16*)(accb + (size_t)Nn * 256);   // [E,32] padded
  bf16*     xbf    = eabf + (size_t)E * 32;              // [Nn,256]
  // transformer-phase view of arena (reused after GAT)
  float* hbuf  = (float*)ar;                             // [T,256] f32
  float* resid = hbuf + (size_t)TMAX * 256;              // [T,256] f32
  bf16*  hbf   = (bf16*)(resid + (size_t)TMAX * 256);    // [T,256]
  bf16*  qbb   = hbf + (size_t)TMAX * 256;
  bf16*  kbb   = qbb + (size_t)TMAX * 256;
  bf16*  vbb   = kbb + (size_t)TMAX * 256;
  bf16*  ctxbf = vbb + (size_t)TMAX * 256;
  bf16*  ffbf  = ctxbf + (size_t)TMAX * 256;             // [T,1024]

  // ------------------------------------------------------------ weight prep
  auto conv = [&](const float* W, bf16* o, int K, int N, int Kpad) {
    k_convW<<<g1d((long)N * Kpad), 256, 0, stream>>>(W, o, K, N, Kpad);
  };
  conv(edge_tr_W, wb_te, 16, 64, 32);
  conv(in_W, wb_in, 160, 256, 160);
  for (int g = 0; g < 2; ++g) {
    conv(lin_W[g], wb_lin[g], 256, 256, 256);
    k_pack_att<<<g1d(256 * 4), 256, 0, stream>>>(lin_W[g], att_src[g], usrc[g], 256);
    k_pack_att<<<g1d(256 * 4), 256, 0, stream>>>(lin_W[g], att_dst[g], udst[g], 256);
    k_pack_att<<<g1d(64 * 4), 256, 0, stream>>>(lin_eW[g], att_edge[g], wed[g], 64);
  }
  for (int l = 0; l < 2; ++l) {
    conv(L[l].Wq, wb_q[l], 256, 256, 256);
    conv(L[l].Wk, wb_k[l], 256, 256, 256);
    conv(L[l].Wv, wb_v[l], 256, 256, 256);
    conv(L[l].Wo, wb_o[l], 256, 256, 256);
    conv(L[l].f1W, wb_f1[l], 256, 1024, 256);
    conv(L[l].f2W, wb_f2[l], 1024, 256, 1024);
  }

  // ------------------------------------------------------------ GAT phase
  k_cvt_pad<<<g1d((long)E * 32), 256, 0, stream>>>(ea, eabf, 16, 32, (long)E * 32);
  k_cvt_pad<<<g1d((long)Nn * 256), 256, 0, stream>>>(xnf, xbf, 256, 256, (long)Nn * 256);
  gemm(stream, eabf, 32, wb_te, te, nullptr, edge_tr_b, E, 64, 32, 32, 1);  // te [E,64]
  k_zero<<<g1d((long)Nn * 64), 256, 0, stream>>>(meanE, (long)Nn * 64);
  k_zero<<<g1d(Nn), 256, 0, stream>>>(deg, Nn);
  k_degsum<<<g1d((long)E * 64), 256, 0, stream>>>(ei + E, te, meanE, deg, E);
  k_mean<<<g1d((long)Nn * 64), 256, 0, stream>>>(meanE, deg, Nn);

  for (int g = 0; g < 2; ++g) {
    gemm(stream, xbf, 256, wb_lin[g], xp, nullptr, nullptr, Nn, 256, 256, 256, 0);
    k_node_att<<<g1d((long)Nn * 4), 256, 0, stream>>>(xnf, usrc[g], udst[g], a_src, a_dst, Nn);
    k_a_edge<<<g1d(E2), 256, 0, stream>>>(te, meanE, wed[g], a_edge, E, E2);
    k_fillu<<<g1d((long)Nn * 4), 256, 0, stream>>>(amax, ENC_NEG_INF, (long)Nn * 4);
    k_zero<<<g1d((long)Nn * 4), 256, 0, stream>>>(denom, (long)Nn * 4);
    k_zero<<<g1d((long)Nn * 256), 256, 0, stream>>>(accb, (long)Nn * 256);
    k_alpha_max<<<g1d((long)E2 * 4), 256, 0, stream>>>(ei, a_src, a_dst, a_edge, amax, E, E2);
    k_denom<<<g1d((long)E2 * 4), 256, 0, stream>>>(ei, a_src, a_dst, a_edge, amax, denom, E, E2);
    k_scatter<<<g1d((long)E2 * 64), 256, 0, stream>>>(ei, a_src, a_dst, a_edge, amax, denom,
                                                      xp, accb, E, E2);
    k_finalize<<<g1d((long)Nn * 64), 256, 0, stream>>>(accb, gbias[g], gout[g], Nn);
  }

  // ------------------------------------------------------------ transformer
  k_comb<<<g1d((long)Bn * 48 * 160), 256, 0, stream>>>(seq, nidx, gout[0], gout[1], combbf, Bn);

  for (int b0 = 0; b0 < Bn; b0 += 256) {
    int nb = (Bn - b0 < 256) ? (Bn - b0) : 256;
    int T  = nb * 48;
    const bf16* cchunk = combbf + (size_t)b0 * 48 * 160;
    gemm(stream, cchunk, 160, wb_in, hbuf, nullptr, in_b, T, 256, 160, 160, 1);
    k_addpos<<<g1d((long)T * 256), 256, 0, stream>>>(hbuf, hbf, pos, T);
    for (int l = 0; l < 2; ++l) {
      gemm(stream, hbf, 256, wb_q[l], nullptr, qbb, L[l].bq, T, 256, 256, 256, 1);
      gemm(stream, hbf, 256, wb_k[l], nullptr, kbb, L[l].bk, T, 256, 256, 256, 1);
      gemm(stream, hbf, 256, wb_v[l], nullptr, vbb, L[l].bv, T, 256, 256, 256, 1);
      k_attn<<<nb * 8, 64, 0, stream>>>(qbb, kbb, vbb, ctxbf);
      gemm(stream, ctxbf, 256, wb_o[l], resid, nullptr, L[l].bo, T, 256, 256, 256, 1);
      k_add_ln<<<g1d((long)T * 32), 256, 0, stream>>>(hbuf, hbf, resid, L[l].ln1s, L[l].ln1b, T);
      gemm(stream, hbf, 256, wb_f1[l], nullptr, ffbf, L[l].f1b, T, 1024, 256, 256, 2);
      gemm(stream, ffbf, 1024, wb_f2[l], resid, nullptr, L[l].f2b, T, 256, 1024, 1024, 1);
      k_add_ln<<<g1d((long)T * 32), 256, 0, stream>>>(hbuf, hbf, resid, L[l].ln2s, L[l].ln2b, T);
    }
    k_out<<<(nb + 7) / 8, 256, 0, stream>>>(hbuf, out_W, out_b, (float*)d_out, nb, b0);
  }
}
